// RecSys_20572893348727
// MI455X (gfx1250) — matrix operand order repaired and checked by reading, not compiled, not run on hardware
//
#include <hip/hip_runtime.h>
#include <hip/hip_bf16.h>
#include <math.h>

typedef _Float16 h16;
typedef __attribute__((ext_vector_type(16))) _Float16 v16h;
typedef __attribute__((ext_vector_type(8)))  _Float16 v8h;
typedef __attribute__((ext_vector_type(4)))  _Float16 v4h;
typedef __attribute__((ext_vector_type(8)))  float    v8f;

#define BB 4096
#define LL 50
#define DD 128
#define HH 64
#define GG 256      // 4*H
#define CC 64
#define KL 192      // D + H  (combined K for LSTM gate GEMM)

// ---- workspace layout (bytes) ----
#define WS_EMBQ   0                                   // [B][128] f32  (zero-init)
#define WS_LWF    (BB*DD*4)                           // LSTM weight frags f16: [2 dir][16 nt][6 ks][32 lane][16 j]
#define WS_LBIAS  (WS_LWF + 2*16*6*32*16*2)           // [2][256] f32 (bih+bhh)
#define WS_SWF    (WS_LBIAS + 2*GG*4)                 // score weight frags f16: [24 nt][12 ks][32][16]
#define WS_SBIAS  (WS_SWF + 24*12*32*16*2)            // [384] f32 conv bias per (c,wi)
#define WS_WCOEF  (WS_SBIAS + 384*4)                  // [384] f32 fc1_w[wi]*fc2_w[c]

// fast activations: native v_exp_f32 + hw reciprocal, short dep chains
__device__ __forceinline__ float fast_sigmoid(float x) {
  return __builtin_amdgcn_rcpf(1.f + __expf(-x));
}
__device__ __forceinline__ float fast_tanh(float x) {
  return 1.f - 2.f * __builtin_amdgcn_rcpf(1.f + __expf(2.f * x));
}
__device__ __forceinline__ v4h cvt4(float4 v) {
  v4h o; o[0] = (h16)v.x; o[1] = (h16)v.y; o[2] = (h16)v.z; o[3] = (h16)v.w; return o;
}

// ------------------------------------------------------------------
// init: zero emb_q + out, fold gate biases, build epilogue coeffs
// ------------------------------------------------------------------
__global__ void k_init(float* embq, float* out, float* lbias, float* sbias, float* wcoef,
                       const float* bihf, const float* bhhf,
                       const float* bihb, const float* bhhb,
                       const float* c1b, const float* c2b, const float* c3b,
                       const float* fc1w, const float* fc2w) {
  int tid = blockIdx.x * blockDim.x + threadIdx.x;
  if (tid < BB * DD) embq[tid] = 0.f;
  if (tid == 0) out[0] = 0.f;
  if (tid < 2 * GG) {
    int dir = tid / GG, g = tid % GG;
    lbias[tid] = dir == 0 ? (bihf[g] + bhhf[g]) : (bihb[g] + bhhb[g]);
  }
  if (tid < 384) {
    int c = tid / 6, wi = tid % 6;
    sbias[tid] = (wi < 3) ? c1b[c] : ((wi < 5) ? c2b[c] : c3b[c]);
    wcoef[tid] = fc1w[wi] * fc2w[c];
  }
}

// ------------------------------------------------------------------
// pack LSTM weights (Wih | Whh per gate row) into f16 B-fragments
// B[K][n] = W[n][K];  b[j] of lane l: N = l%16, K = ks*32 + (l/16)*16 + j
// ------------------------------------------------------------------
__global__ void k_prep_lstm(h16* wfrag,
                            const float* Wihf, const float* Whhf,
                            const float* Wihb, const float* Whhb) {
  int idx = blockIdx.x * blockDim.x + threadIdx.x;          // < 98304
  if (idx >= 2 * 16 * 6 * 32 * 16) return;
  int j    = idx & 15;
  int lane = (idx >> 4) & 31;
  int ks   = (idx >> 9) % 6;
  int nt   = (idx / (512 * 6)) & 15;
  int dir  = idx / (512 * 6 * 16);
  int n = nt * 16 + (lane & 15);
  int K = ks * 32 + (lane >> 4) * 16 + j;
  const float* Wih = dir ? Wihb : Wihf;
  const float* Whh = dir ? Whhb : Whhf;
  float v = (K < DD) ? Wih[n * DD + K] : Whh[n * HH + (K - DD)];
  wfrag[idx] = (h16)v;
}

// ------------------------------------------------------------------
// pack conv head as [384 K][384 N] GEMM weights into f16 B-fragments
// ------------------------------------------------------------------
__global__ void k_prep_score(h16* wfrag,
                             const float* c1w, const float* c2w, const float* c3w) {
  int idx = blockIdx.x * blockDim.x + threadIdx.x;          // < 147456
  if (idx >= 24 * 12 * 32 * 16) return;
  int j    = idx & 15;
  int lane = (idx >> 4) & 31;
  int ks   = (idx >> 9) % 12;
  int nt   = idx / (512 * 12);
  int n = nt * 16 + (lane & 15);
  int c = n / 6, wi = n % 6;
  int K = ks * 32 + (lane >> 4) * 16 + j;
  int d = K & 127, wr = K >> 7;                             // wr in 0..2
  float v = 0.f;
  if (wi < 3)        { if (wr == wi) v = c1w[c * DD + d]; }
  else if (wi < 5)   { int w = wi - 3;
                       if (wr == w)        v = c2w[(c * 2 + 0) * DD + d];
                       else if (wr == w+1) v = c2w[(c * 2 + 1) * DD + d]; }
  else               { v = c3w[(c * 3 + wr) * DD + d]; }
  wfrag[idx] = (h16)v;
}

// ------------------------------------------------------------------
// fused bidirectional LSTM: grid (B/16, dir), 256 threads (8 waves)
// x_t loads are software-pipelined one timestep ahead (b128 loads kept
// in flight across the WMMA + activation + state-update of step t)
// ------------------------------------------------------------------
__global__ void __launch_bounds__(256)
k_lstm(const float* __restrict__ rank_q, const int* __restrict__ qlen,
       const h16* __restrict__ wfrag, const float* __restrict__ gbias,
       float* __restrict__ embq) {
  __shared__ h16   xh[16][KL];      // A staging: K 0..127 = x_t, 128..191 = h (f16)
  __shared__ float cst[16][HH];     // cell state
  __shared__ float gts[16][GG];     // activated gates
  __shared__ int   ln[16];

  const int tid  = threadIdx.x;
  const int wave = tid >> 5, lane = tid & 31;
  const int b0   = blockIdx.x * 16;
  const int dir  = blockIdx.y;

  for (int i = tid; i < 16 * HH; i += 256) {
    cst[i >> 6][i & 63] = 0.f;
    xh[i >> 6][DD + (i & 63)] = (h16)0.f;
  }
  if (tid < 16) ln[tid] = qlen[b0 + tid];

  // this thread's two float4 staging slots (512 float4 items total)
  const int it0 = tid,        m0 = it0 >> 5, d0 = (it0 & 31) << 2;
  const int it1 = tid + 256,  m1 = it1 >> 5, d1 = (it1 & 31) << 2;
  const float* row0 = rank_q + (size_t)(b0 + m0) * LL * DD + d0;
  const float* row1 = rank_q + (size_t)(b0 + m1) * LL * DD + d1;

  // preload time-invariant weight B-fragments into registers (12 x v16h)
  const int nt0 = wave * 2;
  const v16h* wf = (const v16h*)wfrag;
  v16h wB[2][6];
#pragma unroll
  for (int u = 0; u < 2; ++u)
#pragma unroll
    for (int ks = 0; ks < 6; ++ks)
      wB[u][ks] = wf[((dir * 16 + nt0 + u) * 6 + ks) * 32 + lane];

  const float bias0 = gbias[dir * GG + nt0 * 16 + (lane & 15)];
  const float bias1 = gbias[dir * GG + (nt0 + 1) * 16 + (lane & 15)];
  const int gt = nt0 >> 2;          // 0:i 1:f 2:g 3:o

  // prime the pipeline with x(t_first)
  const int tfirst = dir ? (LL - 1) : 0;
  float4 f0 = *(const float4*)(row0 + (size_t)tfirst * DD);
  float4 f1 = *(const float4*)(row1 + (size_t)tfirst * DD);
  float4 n0 = f0, n1 = f1;

  for (int ti = 0; ti < LL; ++ti) {
    const int t = dir ? (LL - 1 - ti) : ti;

    // commit x(t) to LDS (f16); waits on loads issued one step ago
    *(v4h*)&xh[m0][d0] = cvt4(f0);
    *(v4h*)&xh[m1][d1] = cvt4(f1);

    // issue x(t+1) loads now; they stay in flight across this step's work
    if (ti + 1 < LL) {
      const int tn = dir ? (LL - 2 - ti) : (ti + 1);
      n0 = *(const float4*)(row0 + (size_t)tn * DD);
      n1 = *(const float4*)(row1 + (size_t)tn * DD);
    }
    // warm the cache two steps ahead (global_prefetch_b8)
    if (ti + 2 < LL) {
      const int tp = dir ? (LL - 3 - ti) : (ti + 2);
      __builtin_prefetch(row0 + (size_t)tp * DD, 0, 1);
    }
    __syncthreads();

    v8f acc0, acc1;
#pragma unroll
    for (int r = 0; r < 8; ++r) { acc0[r] = bias0; acc1[r] = bias1; }

    const int am = lane & 15;
    const int kh = lane >> 4;
#pragma unroll
    for (int ks = 0; ks < 6; ++ks) {
      const int k0 = ks * 32 + kh * 16;         // 16B-aligned LDS chunks
      const int k1 = ks * 32 + kh * 16 + 8;
      v8h lo = *(const v8h*)&xh[am][k0];
      v8h hi = *(const v8h*)&xh[am][k1];
      v16h a = __builtin_shufflevector(lo, hi, 0,1,2,3,4,5,6,7,8,9,10,11,12,13,14,15);
      acc0 = __builtin_amdgcn_wmma_f32_16x16x32_f16(false, a, false, wB[0][ks], (short)0, acc0, false, false);
      acc1 = __builtin_amdgcn_wmma_f32_16x16x32_f16(false, a, false, wB[1][ks], (short)0, acc1, false, false);
    }

    // per-wave activation (each wave maps to one gate type), write to LDS
#pragma unroll
    for (int r = 0; r < 8; ++r) {
      int M = (lane >> 4) * 8 + r;
      float v0 = acc0[r], v1 = acc1[r];
      if (gt == 2) { v0 = fast_tanh(v0); v1 = fast_tanh(v1); }
      else         { v0 = fast_sigmoid(v0); v1 = fast_sigmoid(v1); }
      gts[M][nt0 * 16 + (lane & 15)]       = v0;
      gts[M][(nt0 + 1) * 16 + (lane & 15)] = v1;
    }
    __syncthreads();

    // state update; capture h only at the gathered timestep (len-1)
    for (int i = tid; i < 16 * HH; i += 256) {
      int m = i >> 6, j = i & 63;
      float ig = gts[m][j], fg = gts[m][64 + j], gg = gts[m][128 + j], og = gts[m][192 + j];
      float c = fg * cst[m][j] + ig * gg;
      float h = og * fast_tanh(c);
      cst[m][j] = c;
      xh[m][DD + j] = (h16)h;
      if (t == ln[m] - 1) embq[(size_t)(b0 + m) * DD + dir * HH + j] = h;
    }
    __syncthreads();

    f0 = n0; f1 = n1;
  }
}

// ------------------------------------------------------------------
// scoring: 16 examples/block; [16x384] x [384x384] WMMA GEMM for both
// low & high matrices; epilogue folds relu+fc1+fc2, bias terms cancel
// in (low - high) so only the data-dependent part is accumulated.
// ------------------------------------------------------------------
__global__ void __launch_bounds__(256)
k_score(const int* __restrict__ r_idx, const int* __restrict__ a_idx,
        const int* __restrict__ acc_idx,
        const float* __restrict__ ru_emb, const float* __restrict__ au_emb,
        const float* __restrict__ embq,
        const h16* __restrict__ wfrag, const float* __restrict__ biasK,
        const float* __restrict__ wcoef, float* __restrict__ out) {
  __shared__ h16 matl[16][384];
  __shared__ h16 math_[16][384];
  __shared__ float red[256];
  const int tid  = threadIdx.x;
  const int wave = tid >> 5, lane = tid & 31;
  const int b0   = blockIdx.x * 16;

  // vectorized gather staging: 2048 float4 items, 8 per thread
  for (int i = tid; i < 16 * 128; i += 256) {
    int m = i >> 7, q = (i & 127) << 2;      // q = 0..508 step 4
    int b = b0 + m;
    float4 v;
    if (q < 128) {
      v = *(const float4*)(ru_emb + (size_t)r_idx[b] * DD + q);
      v4h h = cvt4(v); *(v4h*)&matl[m][q] = h; *(v4h*)&math_[m][q] = h;
    } else if (q < 256) {
      v = *(const float4*)(embq + (size_t)b * DD + (q - 128));
      v4h h = cvt4(v); *(v4h*)&matl[m][q] = h; *(v4h*)&math_[m][q] = h;
    } else if (q < 384) {
      v = *(const float4*)(au_emb + (size_t)a_idx[b] * DD + (q - 256));
      *(v4h*)&matl[m][q] = cvt4(v);
    } else {
      v = *(const float4*)(au_emb + (size_t)acc_idx[b] * DD + (q - 384));
      *(v4h*)&math_[m][q - 128] = cvt4(v);
    }
  }
  __syncthreads();

  const int matsel = (wave * 6) / 24;          // waves 0-3: low, 4-7: high
  const int ntbase = (wave * 6) % 24;
  const h16 (*mat)[384] = matsel ? math_ : matl;

  v8f acc[6];
#pragma unroll
  for (int u = 0; u < 6; ++u)
#pragma unroll
    for (int r = 0; r < 8; ++r) acc[u][r] = 0.f;

  const v16h* wf = (const v16h*)wfrag;
  const int am = lane & 15, kh = lane >> 4;
  for (int ks = 0; ks < 12; ++ks) {
    int k0 = ks * 32 + kh * 16, k1 = ks * 32 + kh * 16 + 8;
    v8h lo = *(const v8h*)&mat[am][k0];
    v8h hi = *(const v8h*)&mat[am][k1];
    v16h a = __builtin_shufflevector(lo, hi, 0,1,2,3,4,5,6,7,8,9,10,11,12,13,14,15);
#pragma unroll
    for (int u = 0; u < 6; ++u) {
      v16h bfr = wf[((ntbase + u) * 12 + ks) * 32 + lane];
      acc[u] = __builtin_amdgcn_wmma_f32_16x16x32_f16(false, a, false, bfr, (short)0, acc[u], false, false);
    }
  }

  float sum = 0.f;
  const float sgn = matsel ? -1.f : 1.f;
#pragma unroll
  for (int u = 0; u < 6; ++u) {
    int n = (ntbase + u) * 16 + (lane & 15);
    float bK = biasK[n], wc = wcoef[n];
#pragma unroll
    for (int r = 0; r < 8; ++r) {
      float s = acc[u][r] + bK;
      s = s > 0.f ? s : 0.f;
      sum += sgn * s * wc;
    }
  }
  red[tid] = sum;
  __syncthreads();
  for (int s = 128; s > 0; s >>= 1) {
    if (tid < s) red[tid] += red[tid + s];
    __syncthreads();
  }
  if (tid == 0) atomicAdd(out, red[0]);
}

// ------------------------------------------------------------------
extern "C" void kernel_launch(void* const* d_in, const int* in_sizes, int n_in,
                              void* d_out, int out_size, void* d_ws, size_t ws_size,
                              hipStream_t stream) {
  const int*   r_idx   = (const int*)d_in[0];
  const int*   a_idx   = (const int*)d_in[1];
  const int*   acc_idx = (const int*)d_in[2];
  const float* rank_q  = (const float*)d_in[3];
  const int*   q_len   = (const int*)d_in[4];
  const float* ru_emb  = (const float*)d_in[5];
  const float* au_emb  = (const float*)d_in[6];
  const float* Wih_f   = (const float*)d_in[7];
  const float* Whh_f   = (const float*)d_in[8];
  const float* bih_f   = (const float*)d_in[9];
  const float* bhh_f   = (const float*)d_in[10];
  const float* Wih_b   = (const float*)d_in[11];
  const float* Whh_b   = (const float*)d_in[12];
  const float* bih_b   = (const float*)d_in[13];
  const float* bhh_b   = (const float*)d_in[14];
  const float* c1w     = (const float*)d_in[15];
  const float* c1b     = (const float*)d_in[16];
  const float* c2w     = (const float*)d_in[17];
  const float* c2b     = (const float*)d_in[18];
  const float* c3w     = (const float*)d_in[19];
  const float* c3b     = (const float*)d_in[20];
  const float* fc1w    = (const float*)d_in[21];
  // d_in[22] = fc1_b, d_in[24] = fc2_b: constant terms cancel in low-high
  const float* fc2w    = (const float*)d_in[23];

  char* ws = (char*)d_ws;
  float* embq  = (float*)(ws + WS_EMBQ);
  h16*   lwf   = (h16*)(ws + WS_LWF);
  float* lbias = (float*)(ws + WS_LBIAS);
  h16*   swf   = (h16*)(ws + WS_SWF);
  float* sbias = (float*)(ws + WS_SBIAS);
  float* wcoef = (float*)(ws + WS_WCOEF);
  float* out   = (float*)d_out;

  k_init<<<(BB * DD + 255) / 256, 256, 0, stream>>>(
      embq, out, lbias, sbias, wcoef,
      bih_f, bhh_f, bih_b, bhh_b, c1b, c2b, c3b, fc1w, fc2w);

  k_prep_lstm<<<(2 * 16 * 6 * 32 * 16) / 256, 256, 0, stream>>>(
      lwf, Wih_f, Whh_f, Wih_b, Whh_b);

  k_prep_score<<<(24 * 12 * 32 * 16) / 256, 256, 0, stream>>>(
      swf, c1w, c2w, c3w);

  k_lstm<<<dim3(BB / 16, 2), 256, 0, stream>>>(rank_q, q_len, lwf, lbias, embq);

  k_score<<<BB / 16, 256, 0, stream>>>(
      r_idx, a_idx, acc_idx, ru_emb, au_emb, embq, swf, sbias, wcoef, out);
}
